// FrequencyInflammationDetector_53309134078402
// MI455X (gfx1250) — matrix operand order repaired and checked by reading, not compiled, not run on hardware
//
#include <hip/hip_runtime.h>

typedef __attribute__((ext_vector_type(16))) __bf16        v16bf;
typedef __attribute__((ext_vector_type(8)))  float         v8f;
typedef __attribute__((ext_vector_type(8)))  unsigned int  v8u;

typedef unsigned short u16;
typedef unsigned int   u32;

// ---------- bf16 helpers ----------
__device__ __forceinline__ u16 f2bf(float f) {
  u32 u = __builtin_bit_cast(u32, f);
  u32 r = u + 0x7FFFu + ((u >> 16) & 1u);   // round-to-nearest-even
  return (u16)(r >> 16);
}

// ---------- WMMA fragment loads (CDNA5 16-bit layouts, wave32) ----------
// A 16x32 (MxK): lane = hi*16+lr, row M = lr.
//   VGPR0..3 = K [hi*8 .. +7], VGPR4..7 = K [16+hi*8 .. +7]  -> two b128 loads
__device__ __forceinline__ v16bf load_a_frag(const u16* row, int k0, int hi) {
  const uint4* p0 = (const uint4*)(row + k0 + hi * 8);
  const uint4* p1 = (const uint4*)(row + k0 + 16 + hi * 8);
  uint4 x0 = *p0, x1 = *p1;
  v8u u;
  u[0] = x0.x; u[1] = x0.y; u[2] = x0.z; u[3] = x0.w;
  u[4] = x1.x; u[5] = x1.y; u[6] = x1.z; u[7] = x1.w;
  return __builtin_bit_cast(v16bf, u);
}
// B 32x16 (KxN): lane = hi*16+lr, col N = lr. VGPR0..3 = K [hi*8 .. +7], VGPR4..7 = K [16+hi*8 .. +7].
// LDS tile stored [N][K] -> two b128 loads per fragment.
__device__ __forceinline__ v16bf load_b_frag(const u16* row, int hi) {
  const uint4* p0 = (const uint4*)(row + hi * 8);
  const uint4* p1 = (const uint4*)(row + 16 + hi * 8);
  uint4 x0 = *p0, x1 = *p1;
  v8u u;
  u[0] = x0.x; u[1] = x0.y; u[2] = x0.z; u[3] = x0.w;
  u[4] = x1.x; u[5] = x1.y; u[6] = x1.z; u[7] = x1.w;
  return __builtin_bit_cast(v16bf, u);
}

__device__ __forceinline__ v8f wmma_bf16(v16bf a, v16bf b, v8f c) {
  return __builtin_amdgcn_wmma_f32_16x16x32_bf16(
      false, a, false, b, (short)0, c, false, false);
}

// =====================================================================
// Weight prep: OIHW f32 -> bf16 [COUT][KPAD], k = tap*CIN + cin, 0-padded.
// =====================================================================
template <int CIN, int COUT, int KPAD>
__global__ void __launch_bounds__(256) wprep3x3(const float* __restrict__ src,
                                                u16* __restrict__ dst) {
  int idx = blockIdx.x * 256 + threadIdx.x;
  if (idx >= COUT * KPAD) return;
  int o = idx / KPAD, k = idx - o * KPAD;
  u16 v = 0;
  if (k < CIN * 9) {
    int tap = k / CIN, ci = k - tap * CIN;
    v = f2bf(src[(o * CIN + ci) * 9 + tap]);
  }
  dst[idx] = v;
}

__global__ void __launch_bounds__(256) wprep1x1(const float* __restrict__ src,
                                                u16* __restrict__ dst, int n) {
  int idx = blockIdx.x * 256 + threadIdx.x;
  if (idx < n) dst[idx] = f2bf(src[idx]);
}

// =====================================================================
// Kernel 1: conv1x1(3->16)+ReLU, conv1x1(16->3), Haar DWT.
// x (16,3,512,512) f32 NCHW -> low NHWC bf16 (3ch), high NHWC bf16 (9ch)
// =====================================================================
__global__ void __launch_bounds__(256) front_kernel(
    const float* __restrict__ x,
    const float* __restrict__ cw1, const float* __restrict__ cb1,
    const float* __restrict__ cw2, const float* __restrict__ cb2,
    u16* __restrict__ low, u16* __restrict__ high) {
  __shared__ float sw1[48], sb1[16], sw2[48], sb2[3];
  int tid = threadIdx.x;
  if (tid < 48) sw1[tid] = cw1[tid];
  if (tid < 16) sb1[tid] = cb1[tid];
  if (tid >= 64 && tid < 112) sw2[tid - 64] = cw2[tid - 64];
  if (tid >= 128 && tid < 131) sb2[tid - 128] = cb2[tid - 128];
  __syncthreads();

  long m = (long)blockIdx.x * 256 + tid;      // (b, ho, wo) over 16*256*256
  int b  = (int)(m >> 16);
  int hw = (int)(m & 65535);
  int ho = hw >> 8, wo = hw & 255;

  float hsv[4][3];
#pragma unroll
  for (int p = 0; p < 4; ++p) {
    int hi2 = 2 * ho + (p >> 1), wi = 2 * wo + (p & 1);
    float xi[3];
#pragma unroll
    for (int i = 0; i < 3; ++i)
      xi[i] = x[(((long)b * 3 + i) << 18) + ((long)hi2 << 9) + wi];
    float o0 = sb2[0], o1 = sb2[1], o2 = sb2[2];
#pragma unroll
    for (int j = 0; j < 16; ++j) {
      float h = sb1[j] + sw1[j * 3 + 0] * xi[0] + sw1[j * 3 + 1] * xi[1] +
                sw1[j * 3 + 2] * xi[2];
      h = fmaxf(h, 0.f);
      o0 += sw2[0 * 16 + j] * h;
      o1 += sw2[1 * 16 + j] * h;
      o2 += sw2[2 * 16 + j] * h;
    }
    hsv[p][0] = o0; hsv[p][1] = o1; hsv[p][2] = o2;
  }

  long pix = ((long)b << 16) + hw;
#pragma unroll
  for (int c = 0; c < 3; ++c) {
    float a = hsv[0][c], bq = hsv[1][c], cq = hsv[2][c], dq = hsv[3][c];
    low [pix * 3 + c]         = f2bf((a + bq + cq + dq) * 0.5f);
    high[pix * 9 + c * 3 + 0] = f2bf((a + bq - cq - dq) * 0.5f);
    high[pix * 9 + c * 3 + 1] = f2bf((a - bq + cq - dq) * 0.5f);
    high[pix * 9 + c * 3 + 2] = f2bf((a - bq - cq + dq) * 0.5f);
  }
}

// =====================================================================
// Small-CIN 3x3 conv (CIN=3 or 9) + bias + BN + ReLU, WMMA bf16.
// NHWC in/out, K_TILE=32, element-wise im2col staging (few GFLOP total).
// =====================================================================
template <int CIN, int KPAD, int COUT>
__global__ void __launch_bounds__(256) conv3x3_small(
    const u16* __restrict__ in, const u16* __restrict__ wb,
    const float* __restrict__ bias, const float* __restrict__ gam,
    const float* __restrict__ bet, const float* __restrict__ mu,
    const float* __restrict__ var, u16* __restrict__ out) {
  constexpr int KTOT = CIN * 9;
  constexpr int KTILES = KPAD / 32;
  constexpr int WN = COUT / 2;
  constexpr int NFRAG = WN / 16;
  __shared__ u16 lA[128 * 32];
  __shared__ u16 lB[COUT * 32];

  int tid = threadIdx.x, lane = tid & 31, wave = tid >> 5;
  int wm = wave & 3, wn = wave >> 2;
  int lr = lane & 15, hi = lane >> 4;
  long mbase = (long)blockIdx.x * 128;

  v8f acc[2][NFRAG];
#pragma unroll
  for (int mi = 0; mi < 2; ++mi)
#pragma unroll
    for (int ni = 0; ni < NFRAG; ++ni)
#pragma unroll
      for (int r = 0; r < 8; ++r) acc[mi][ni][r] = 0.f;

#pragma unroll
  for (int kt = 0; kt < KTILES; ++kt) {
    int k0 = kt * 32;
    // A tile: m-major iteration -> coalesced global reads
#pragma unroll
    for (int it = 0; it < 16; ++it) {
      int idx = it * 256 + tid;
      int ml = idx & 127, kk = idx >> 7;
      int k = k0 + kk;
      u16 v = 0;
      if (k < KTOT) {
        int tap = k / CIN, ci = k - tap * CIN;
        int rr = tap / 3, ss = tap - rr * 3;
        long mg = mbase + ml;
        int b = (int)(mg >> 16), hw = (int)(mg & 65535);
        int h = hw >> 8, w = hw & 255;
        int hh = h + rr - 1, ww = w + ss - 1;
        if (hh >= 0 && hh < 256 && ww >= 0 && ww < 256)
          v = in[(((long)b << 16) + (hh << 8) + ww) * CIN + ci];
      }
      lA[ml * 32 + kk] = v;
    }
    // B tile: contiguous bf16 -> b128 copies
#pragma unroll
    for (int it = 0; it < (COUT * 32) / 2048; ++it) {
      int idx = it * 256 + tid;
      int nl = idx >> 2, part = idx & 3;
      *(uint4*)&lB[nl * 32 + part * 8] =
          *(const uint4*)(wb + nl * KPAD + k0 + part * 8);
    }
    __syncthreads();

    v16bf af[2], bf[NFRAG];
#pragma unroll
    for (int mi = 0; mi < 2; ++mi)
      af[mi] = load_a_frag(&lA[(wm * 32 + mi * 16 + lr) * 32], 0, hi);
#pragma unroll
    for (int ni = 0; ni < NFRAG; ++ni)
      bf[ni] = load_b_frag(&lB[(wn * WN + ni * 16 + lr) * 32], hi);
#pragma unroll
    for (int mi = 0; mi < 2; ++mi)
#pragma unroll
      for (int ni = 0; ni < NFRAG; ++ni)
        acc[mi][ni] = wmma_bf16(af[mi], bf[ni], acc[mi][ni]);
    __syncthreads();
  }

#pragma unroll
  for (int ni = 0; ni < NFRAG; ++ni) {
    int n = wn * WN + ni * 16 + lr;
    float sc = gam[n] * rsqrtf(var[n] + 1e-5f);
    float sh = bet[n] - mu[n] * sc;
    float bb = bias[n];
#pragma unroll
    for (int mi = 0; mi < 2; ++mi) {
#pragma unroll
      for (int r = 0; r < 8; ++r) {
        float v = fmaxf((acc[mi][ni][r] + bb) * sc + sh, 0.f);
        int ml = wm * 32 + mi * 16 + hi * 8 + r;
        long mg = mbase + ml;
        out[((((long)(mg >> 16)) << 16) + (mg & 65535)) * COUT + n] = f2bf(v);
      }
    }
  }
}

// =====================================================================
// Big 3x3 conv, CIN=64 -> COUT, K_TILE = 64 = exactly one filter tap.
// Fully unrolled tap loop => compile-time (r,s), pure b128 staging.
// NHWC bf16 in; NHWC bf16 out into CTOT_OUT-channel buffer at COFF.
// =====================================================================
template <int COUT, int CTOT_OUT, int COFF>
__global__ void __launch_bounds__(256) conv3x3_c64(
    const u16* __restrict__ in, const u16* __restrict__ wb,  // [COUT][576]
    const float* __restrict__ bias, const float* __restrict__ gam,
    const float* __restrict__ bet, const float* __restrict__ mu,
    const float* __restrict__ var, u16* __restrict__ out) {
  constexpr int WN = COUT / 2;
  constexpr int NFRAG = WN / 16;
  __shared__ u16 lA[128 * 64];
  __shared__ u16 lB[COUT * 64];

  int tid = threadIdx.x, lane = tid & 31, wave = tid >> 5;
  int wm = wave & 3, wn = wave >> 2;
  int lr = lane & 15, hi = lane >> 4;
  long mbase = (long)blockIdx.x * 128;

  v8f acc[2][NFRAG];
#pragma unroll
  for (int mi = 0; mi < 2; ++mi)
#pragma unroll
    for (int ni = 0; ni < NFRAG; ++ni)
#pragma unroll
      for (int r = 0; r < 8; ++r) acc[mi][ni][r] = 0.f;

#pragma unroll
  for (int tap = 0; tap < 9; ++tap) {
    const int rr = tap / 3, ss = tap - rr * 3;   // compile-time after unroll
    // ---- A tile: 128 rows x 64ch, one tap; 4 b128 copies / thread ----
#pragma unroll
    for (int it = 0; it < 4; ++it) {
      int idx = it * 256 + tid;
      int row = idx >> 3, part = idx & 7;
      long mg = mbase + row;
      int b = (int)(mg >> 16), hw = (int)(mg & 65535);
      int h = hw >> 8, w = hw & 255;
      int hh = h + rr - 1, ww = w + ss - 1;
      uint4 val = {0u, 0u, 0u, 0u};
      if (hh >= 0 && hh < 256 && ww >= 0 && ww < 256)
        val = *(const uint4*)(in + (((((long)b << 16) + (hh << 8) + ww)) << 6) +
                              part * 8);
      *(uint4*)&lA[row * 64 + part * 8] = val;
    }
    // ---- B tile: COUT rows x 64, contiguous b128 copies ----
#pragma unroll
    for (int it = 0; it < (COUT * 64) / 2048; ++it) {
      int idx = it * 256 + tid;
      int nl = idx >> 3, part = idx & 7;
      *(uint4*)&lB[nl * 64 + part * 8] =
          *(const uint4*)(wb + nl * 576 + tap * 64 + part * 8);
      if (tap < 8)  // prefetch next tap's weights (global_prefetch_b8)
        __builtin_prefetch(wb + nl * 576 + (tap + 1) * 64 + part * 8, 0, 0);
    }
    __syncthreads();

#pragma unroll
    for (int ks = 0; ks < 2; ++ks) {
      v16bf af[2], bf[NFRAG];
#pragma unroll
      for (int mi = 0; mi < 2; ++mi)
        af[mi] = load_a_frag(&lA[(wm * 32 + mi * 16 + lr) * 64], ks * 32, hi);
#pragma unroll
      for (int ni = 0; ni < NFRAG; ++ni)
        bf[ni] = load_b_frag(&lB[(wn * WN + ni * 16 + lr) * 64 + ks * 32], hi);
#pragma unroll
      for (int mi = 0; mi < 2; ++mi)
#pragma unroll
        for (int ni = 0; ni < NFRAG; ++ni)
          acc[mi][ni] = wmma_bf16(af[mi], bf[ni], acc[mi][ni]);
    }
    __syncthreads();
  }

#pragma unroll
  for (int ni = 0; ni < NFRAG; ++ni) {
    int n = wn * WN + ni * 16 + lr;
    float sc = gam[n] * rsqrtf(var[n] + 1e-5f);
    float sh = bet[n] - mu[n] * sc;
    float bb = bias[n];
#pragma unroll
    for (int mi = 0; mi < 2; ++mi) {
#pragma unroll
      for (int r = 0; r < 8; ++r) {
        float v = fmaxf((acc[mi][ni][r] + bb) * sc + sh, 0.f);
        int ml = wm * 32 + mi * 16 + hi * 8 + r;
        long mg = mbase + ml;
        long pix = (((long)(mg >> 16)) << 16) + (mg & 65535);
        out[pix * CTOT_OUT + COFF + n] = f2bf(v);
      }
    }
  }
}

// =====================================================================
// Fused 1x1 convs: [Mx256]x[256x128] -> BN/ReLU -> LDS bf16
//                  [Mx128]x[128x64]  -> BN/ReLU -> coalesced NCHW f32
// =====================================================================
__global__ void __launch_bounds__(256) fuse1x1_kernel(
    const u16* __restrict__ F,    // NHWC bf16, 256 channels
    const u16* __restrict__ wb1, const float* __restrict__ b1,
    const float* __restrict__ g1, const float* __restrict__ be1,
    const float* __restrict__ m1, const float* __restrict__ v1,
    const u16* __restrict__ wb2, const float* __restrict__ b2,
    const float* __restrict__ g2, const float* __restrict__ be2,
    const float* __restrict__ m2, const float* __restrict__ v2,
    float* __restrict__ out) {
  __shared__ u16 smem[32768];        // 64 KB
  u16* lA    = smem;                 // 128 x 64
  u16* lB    = smem + 8192;          // 128 x 64
  u16* inter = smem + 16384;         // 128 x 128 bf16 intermediate

  int tid = threadIdx.x, lane = tid & 31, wave = tid >> 5;
  int wm = wave & 3, wn = wave >> 2;
  int lr = lane & 15, hi = lane >> 4;
  long mbase = (long)blockIdx.x * 128;

  // -------- stage 1: 256 -> 128, K_TILE = 64 --------
  {
    v8f acc[2][4];
#pragma unroll
    for (int mi = 0; mi < 2; ++mi)
#pragma unroll
      for (int ni = 0; ni < 4; ++ni)
#pragma unroll
        for (int r = 0; r < 8; ++r) acc[mi][ni][r] = 0.f;

#pragma unroll
    for (int kt = 0; kt < 4; ++kt) {
      int k0 = kt * 64;
#pragma unroll
      for (int it = 0; it < 4; ++it) {
        int idx = it * 256 + tid;
        int row = idx >> 3, part = idx & 7;
        long mg = mbase + row;
        long pix = (((long)(mg >> 16)) << 16) + (mg & 65535);
        *(uint4*)&lA[row * 64 + part * 8] =
            *(const uint4*)(F + (pix << 8) + k0 + part * 8);
      }
#pragma unroll
      for (int it = 0; it < 4; ++it) {
        int idx = it * 256 + tid;
        int nl = idx >> 3, part = idx & 7;
        *(uint4*)&lB[nl * 64 + part * 8] =
            *(const uint4*)(wb1 + (nl << 8) + k0 + part * 8);
      }
      __syncthreads();
#pragma unroll
      for (int ks = 0; ks < 2; ++ks) {
        v16bf af[2], bf[4];
#pragma unroll
        for (int mi = 0; mi < 2; ++mi)
          af[mi] = load_a_frag(&lA[(wm * 32 + mi * 16 + lr) * 64], ks * 32, hi);
#pragma unroll
        for (int ni = 0; ni < 4; ++ni)
          bf[ni] = load_b_frag(&lB[(wn * 64 + ni * 16 + lr) * 64 + ks * 32], hi);
#pragma unroll
        for (int mi = 0; mi < 2; ++mi)
#pragma unroll
          for (int ni = 0; ni < 4; ++ni)
            acc[mi][ni] = wmma_bf16(af[mi], bf[ni], acc[mi][ni]);
      }
      __syncthreads();
    }

#pragma unroll
    for (int ni = 0; ni < 4; ++ni) {
      int n = wn * 64 + ni * 16 + lr;
      float sc = g1[n] * rsqrtf(v1[n] + 1e-5f);
      float sh = be1[n] - m1[n] * sc;
      float bb = b1[n];
#pragma unroll
      for (int mi = 0; mi < 2; ++mi) {
#pragma unroll
        for (int r = 0; r < 8; ++r) {
          float v = fmaxf((acc[mi][ni][r] + bb) * sc + sh, 0.f);
          int ml = wm * 32 + mi * 16 + hi * 8 + r;
          inter[ml * 128 + n] = f2bf(v);
        }
      }
    }
    __syncthreads();
  }

  // -------- stage 2: 128 -> 64, K_TILE = 64, A from LDS inter --------
  {
    v8f acc[2][2];
#pragma unroll
    for (int mi = 0; mi < 2; ++mi)
#pragma unroll
      for (int ni = 0; ni < 2; ++ni)
#pragma unroll
        for (int r = 0; r < 8; ++r) acc[mi][ni][r] = 0.f;

#pragma unroll
    for (int kt = 0; kt < 2; ++kt) {
      int k0 = kt * 64;
#pragma unroll
      for (int it = 0; it < 2; ++it) {
        int idx = it * 256 + tid;      // 512 uint4 = 64 rows x 8 parts
        int nl = idx >> 3, part = idx & 7;
        *(uint4*)&lB[nl * 64 + part * 8] =
            *(const uint4*)(wb2 + (nl << 7) + k0 + part * 8);
      }
      __syncthreads();
#pragma unroll
      for (int ks = 0; ks < 2; ++ks) {
        v16bf af[2], bf[2];
#pragma unroll
        for (int mi = 0; mi < 2; ++mi)
          af[mi] = load_a_frag(&inter[(wm * 32 + mi * 16 + lr) * 128],
                               k0 + ks * 32, hi);
#pragma unroll
        for (int ni = 0; ni < 2; ++ni)
          bf[ni] = load_b_frag(&lB[(wn * 32 + ni * 16 + lr) * 64 + ks * 32], hi);
#pragma unroll
        for (int mi = 0; mi < 2; ++mi)
#pragma unroll
          for (int ni = 0; ni < 2; ++ni)
            acc[mi][ni] = wmma_bf16(af[mi], bf[ni], acc[mi][ni]);
      }
      __syncthreads();
    }

    // epilogue 2: BN/ReLU -> LDS transpose buffer -> coalesced NCHW stores
    float* tbuf = (float*)smem;   // 128 x 64 f32 over lA+lB region (32 KB)
#pragma unroll
    for (int ni = 0; ni < 2; ++ni) {
      int n = wn * 32 + ni * 16 + lr;
      float sc = g2[n] * rsqrtf(v2[n] + 1e-5f);
      float sh = be2[n] - m2[n] * sc;
      float bb = b2[n];
#pragma unroll
      for (int mi = 0; mi < 2; ++mi) {
#pragma unroll
        for (int r = 0; r < 8; ++r) {
          float v = fmaxf((acc[mi][ni][r] + bb) * sc + sh, 0.f);
          int ml = wm * 32 + mi * 16 + hi * 8 + r;
          tbuf[ml * 64 + n] = v;
        }
      }
    }
    __syncthreads();
#pragma unroll
    for (int it = 0; it < 32; ++it) {
      int idx = it * 256 + tid;     // 8192 = 64 ch x 128 pixels
      int n = idx >> 7, ml = idx & 127;
      long mg = mbase + ml;
      int b = (int)(mg >> 16), hw = (int)(mg & 65535);
      out[(((long)b * 64 + n) << 16) + hw] = tbuf[ml * 64 + n];
    }
  }
}

// =====================================================================
// Launch
// =====================================================================
extern "C" void kernel_launch(void* const* d_in, const int* in_sizes, int n_in,
                              void* d_out, int out_size, void* d_ws,
                              size_t ws_size, hipStream_t stream) {
  (void)in_sizes; (void)n_in; (void)out_size; (void)ws_size;
  const float* x   = (const float*)d_in[0];
  const float* cw1 = (const float*)d_in[1];
  const float* cb1 = (const float*)d_in[2];
  const float* cw2 = (const float*)d_in[3];
  const float* cb2 = (const float*)d_in[4];
  const float* lw1 = (const float*)d_in[5];
  const float* lb1 = (const float*)d_in[6];
  const float* lg1 = (const float*)d_in[7];
  const float* lbe1= (const float*)d_in[8];
  const float* lm1 = (const float*)d_in[9];
  const float* lv1 = (const float*)d_in[10];
  const float* lw2 = (const float*)d_in[11];
  const float* lb2 = (const float*)d_in[12];
  const float* lg2 = (const float*)d_in[13];
  const float* lbe2= (const float*)d_in[14];
  const float* lm2 = (const float*)d_in[15];
  const float* lv2 = (const float*)d_in[16];
  const float* hw1 = (const float*)d_in[17];
  const float* hb1 = (const float*)d_in[18];
  const float* hg1 = (const float*)d_in[19];
  const float* hbe1= (const float*)d_in[20];
  const float* hm1 = (const float*)d_in[21];
  const float* hv1 = (const float*)d_in[22];
  const float* hw2 = (const float*)d_in[23];
  const float* hb2 = (const float*)d_in[24];
  const float* hg2 = (const float*)d_in[25];
  const float* hbe2= (const float*)d_in[26];
  const float* hm2 = (const float*)d_in[27];
  const float* hv2 = (const float*)d_in[28];
  const float* fw1 = (const float*)d_in[29];
  const float* fb1 = (const float*)d_in[30];
  const float* fg1 = (const float*)d_in[31];
  const float* fbe1= (const float*)d_in[32];
  const float* fm1 = (const float*)d_in[33];
  const float* fv1 = (const float*)d_in[34];
  const float* fw2 = (const float*)d_in[35];
  const float* fb2 = (const float*)d_in[36];
  const float* fg2 = (const float*)d_in[37];
  const float* fbe2= (const float*)d_in[38];
  const float* fm2 = (const float*)d_in[39];
  const float* fv2 = (const float*)d_in[40];

  const size_t NPIX = (size_t)16 * 256 * 256;
  u16* low  = (u16*)d_ws;            // NHWC 3ch
  u16* high = low  + 3   * NPIX;     // NHWC 9ch
  u16* lf1  = high + 9   * NPIX;     // NHWC 64ch
  u16* hf1  = lf1  + 64  * NPIX;     // NHWC 64ch
  u16* F    = hf1  + 64  * NPIX;     // NHWC 256ch (lf2 | hf2 concat)
  u16* wbL1 = F    + 256 * NPIX;     // [64][32]
  u16* wbH1 = wbL1 + 64 * 32;        // [64][96]
  u16* wbL2 = wbH1 + 64 * 96;        // [128][576]
  u16* wbH2 = wbL2 + 128 * 576;      // [128][576]
  u16* wbF1 = wbH2 + 128 * 576;      // [128][256]
  u16* wbF2 = wbF1 + 128 * 256;      // [64][128]
  float* out = (float*)d_out;

  const int MBLKS = (int)(NPIX / 128);  // 8192

  // weight prep (tiny)
  wprep3x3<3, 64, 32>  <<<8,   256, 0, stream>>>(lw1, wbL1);
  wprep3x3<9, 64, 96>  <<<24,  256, 0, stream>>>(hw1, wbH1);
  wprep3x3<64, 128, 576><<<288, 256, 0, stream>>>(lw2, wbL2);
  wprep3x3<64, 128, 576><<<288, 256, 0, stream>>>(hw2, wbH2);
  wprep1x1<<<128, 256, 0, stream>>>(fw1, wbF1, 128 * 256);
  wprep1x1<<<32,  256, 0, stream>>>(fw2, wbF2, 64 * 128);

  front_kernel<<<(int)(NPIX / 256), 256, 0, stream>>>(x, cw1, cb1, cw2, cb2,
                                                      low, high);
  conv3x3_small<3, 32, 64>
      <<<MBLKS, 256, 0, stream>>>(low, wbL1, lb1, lg1, lbe1, lm1, lv1, lf1);
  conv3x3_small<9, 96, 64>
      <<<MBLKS, 256, 0, stream>>>(high, wbH1, hb1, hg1, hbe1, hm1, hv1, hf1);
  conv3x3_c64<128, 256, 0>
      <<<MBLKS, 256, 0, stream>>>(lf1, wbL2, lb2, lg2, lbe2, lm2, lv2, F);
  conv3x3_c64<128, 256, 128>
      <<<MBLKS, 256, 0, stream>>>(hf1, wbH2, hb2, hg2, hbe2, hm2, hv2, F);
  fuse1x1_kernel<<<MBLKS, 256, 0, stream>>>(F, wbF1, fb1, fg1, fbe1, fm1, fv1,
                                            wbF2, fb2, fg2, fbe2, fm2, fv2,
                                            out);
}
